// ScaleAwareLogRatioConditionalGraphNetwork_5428838662516
// MI455X (gfx1250) — compile-verified, matrix operand order checked by reading, not run on hardware
//
#include <hip/hip_runtime.h>
#include <hip/hip_bf16.h>

typedef __attribute__((ext_vector_type(16))) _Float16 v16h;
typedef __attribute__((ext_vector_type(8)))  _Float16 v8h;
typedef __attribute__((ext_vector_type(8)))  float    v8f;

#define HID 128
#define MT  32          // rows per workgroup tile
#define AS  520         // LDS A-stage row stride in halfs (max K1=512, +8 pad vs banks)
#define HS  136         // LDS hidden row stride in halfs (128 + 8 pad)

// ---------------------------------------------------------------------------
// Weight convert+transpose: W[K,N] f32 -> Wt[n*K+k] f16
// ---------------------------------------------------------------------------
__global__ void k_wt(const float* __restrict__ W, _Float16* __restrict__ Wt, int K, int N) {
    int t = blockIdx.x * 256 + threadIdx.x;
    if (t < K * N) {
        int k = t / N, n = t % N;
        Wt[(long)n * K + k] = (_Float16)W[t];
    }
}

// ---------------------------------------------------------------------------
// edge_batch = batch[row[e]]; cnt[row[e]] += 1
// ---------------------------------------------------------------------------
__global__ void k_edge_batch_cnt(const int* __restrict__ row, const int* __restrict__ batch,
                                 int E, int* __restrict__ eb, float* __restrict__ cnt) {
    int e = blockIdx.x * 256 + threadIdx.x;
    if (e < E) {
        int r = row[e];
        eb[e] = batch[r];
        atomicAdd(&cnt[r], 1.0f);
    }
}

// ---------------------------------------------------------------------------
// Global path (B=8): cond_enc, scale_enc, u_enc. Single block, 128 threads.
// ---------------------------------------------------------------------------
__global__ __launch_bounds__(128)
void k_u_path(const float* cond, const float* scal,
              const float* cw1, const float* cb1, const float* cw2, const float* cb2,
              const float* sw1, const float* sb1, const float* sw2, const float* sb2,
              const float* uw1, const float* ub1, const float* uw2, const float* ub2,
              int B, float* u32, _Float16* u16) {
    __shared__ float cenc[8 * HID];
    __shared__ float senc[8 * HID];
    __shared__ float tmp[HID];
    if (B > 8) B = 8;
    int j = threadIdx.x;
    for (int b = 0; b < B; ++b) {
        float a = cb1[j];
        for (int k = 0; k < 16; ++k) a += cond[b * 16 + k] * cw1[k * HID + j];
        tmp[j] = fmaxf(a, 0.0f);
        __syncthreads();
        float o = cb2[j];
        for (int k = 0; k < HID; ++k) o += tmp[k] * cw2[k * HID + j];
        cenc[b * HID + j] = o;
        __syncthreads();
        a = sb1[j];
        for (int k = 0; k < 4; ++k) a += scal[b * 4 + k] * sw1[k * HID + j];
        tmp[j] = fmaxf(a, 0.0f);
        __syncthreads();
        o = sb2[j];
        for (int k = 0; k < HID; ++k) o += tmp[k] * sw2[k * HID + j];
        senc[b * HID + j] = o;
        __syncthreads();
    }
    for (int b = 0; b < B; ++b) {
        float a = ub1[j];
        for (int k = 0; k < HID; ++k) a += cenc[b * HID + k] * uw1[k * HID + j];
        for (int k = 0; k < HID; ++k) a += senc[b * HID + k] * uw1[(HID + k) * HID + j];
        tmp[j] = fmaxf(a, 0.0f);
        __syncthreads();
        float o = ub2[j];
        for (int k = 0; k < HID; ++k) o += tmp[k] * uw2[k * HID + j];
        u32[b * HID + j] = o;
        u16[b * HID + j] = (_Float16)o;
        __syncthreads();
    }
}

// ---------------------------------------------------------------------------
// Fused 2-layer MLP on 32-row tiles. 256 threads = 8 waves; wave w owns output
// columns [16w,16w+16) and keeps TWO accumulators (rows 0-15 / 16-31) so each
// B (weight) load feeds two WMMAs.
//
// Layer1 (WMMA path): the gathered/concatenated A tile [32, K1] (up to 4
// indexed f16 sources of width 128) is cooperatively staged to LDS ONCE
// (instead of 8 waves redundantly gathering it from global), then consumed
// via ds_load_b128 in the ISA's 16-bit A layout:
//   lane l (m=l%16, b8=(l>=16)?8:0) holds K = {k0+b8..+7, k0+16+b8..+7}.
// Thin layer1 (din<=32, f32 input) is a VALU path straight into LDS hidden.
// Hidden (ReLU, f16) lives in LDS (padded stride). Layer2: WMMA with bias
// (+ optional residual, f32/f16 outputs), or VALU path for tiny dout.
// ---------------------------------------------------------------------------
__global__ __launch_bounds__(256)
void fused_mlp2(int M,
                const float* thin_in, int din, const float* w1_thin,
                const _Float16* s0, const int* i0,
                const _Float16* s1, const int* i1,
                const _Float16* s2, const int* i2,
                const _Float16* s3, const int* i3,
                int nsrc,
                const _Float16* w1t,   // [128][nsrc*128] (transposed)
                const float* b1,
                const _Float16* w2t,   // [128][128] (transposed), WMMA layer2
                const float* b2,
                const float* w2_naive, int dout_naive,   // VALU layer2 if >0
                const float* res32,
                float* out32, _Float16* out16) {
    __shared__ _Float16 ain[MT * AS];
    __shared__ _Float16 hid[MT * HS];
    const int m0   = blockIdx.x * MT;
    const int t    = threadIdx.x;
    const int wave = t >> 5;
    const int lane = t & 31;
    const int hi   = lane >> 4;      // half-wave select
    const int b8   = hi * 8;
    const int rloc = lane & 15;

    if (thin_in) {
        // ---- layer1 VALU (thin input) ----
        int j  = t & 127;
        int rg = t >> 7;
        for (int rr = 0; rr < 16; ++rr) {
            int ml = rg * 16 + rr;
            int m  = m0 + ml;
            int mm = m < M ? m : M - 1;
            float acc = b1[j];
            for (int k = 0; k < din; ++k)
                acc += thin_in[(long)mm * din + k] * w1_thin[k * HID + j];
            hid[ml * HS + j] = (_Float16)fmaxf(acc, 0.0f);
        }
    } else {
        const int K1  = nsrc * HID;
        const int CPR = K1 >> 3;              // 16B chunks per row
        // ---- stage gathered A tile [MT, K1] into LDS (once per block) ----
        for (int cch = t; cch < MT * CPR; cch += 256) {
            int r   = cch / CPR;
            int koh = (cch - r * CPR) * 8;    // half offset within row
            int s   = koh >> 7;
            int wo  = koh & 127;
            const _Float16* sp = (s == 0) ? s0 : (s == 1) ? s1 : (s == 2) ? s2 : s3;
            const int*      ip = (s == 0) ? i0 : (s == 1) ? i1 : (s == 2) ? i2 : i3;
            int m  = m0 + r;
            int mm = m < M ? m : M - 1;
            int ri = ip ? ip[mm] : mm;
            *(v8h*)(ain + r * AS + koh) = *(const v8h*)(sp + (long)ri * HID + wo);
        }
        __syncthreads();
        // ---- layer1 WMMA: two row-halves share each B load ----
        v8f c0 = {}, c1 = {};
        const int n = wave * 16 + rloc;
        const _Float16* brow = w1t + (long)n * K1;
        const _Float16* ar0  = ain + rloc * AS;
        const _Float16* ar1  = ain + (16 + rloc) * AS;
        for (int k0 = 0; k0 < K1; k0 += 32) {
            v8h a0lo = *(const v8h*)(ar0 + k0 + b8);
            v8h a0hi = *(const v8h*)(ar0 + k0 + 16 + b8);
            v8h a1lo = *(const v8h*)(ar1 + k0 + b8);
            v8h a1hi = *(const v8h*)(ar1 + k0 + 16 + b8);
            v8h blo  = *(const v8h*)(brow + k0 + b8);
            v8h bhi  = *(const v8h*)(brow + k0 + 16 + b8);
            v16h a0 = __builtin_shufflevector(a0lo, a0hi, 0,1,2,3,4,5,6,7,8,9,10,11,12,13,14,15);
            v16h a1 = __builtin_shufflevector(a1lo, a1hi, 0,1,2,3,4,5,6,7,8,9,10,11,12,13,14,15);
            v16h b  = __builtin_shufflevector(blo,  bhi,  0,1,2,3,4,5,6,7,8,9,10,11,12,13,14,15);
            c0 = __builtin_amdgcn_wmma_f32_16x16x32_f16(false, a0, false, b, (short)0, c0, false, false);
            c1 = __builtin_amdgcn_wmma_f32_16x16x32_f16(false, a1, false, b, (short)0, c1, false, false);
        }
        const float bias = b1[n];
#pragma unroll
        for (int r = 0; r < 8; ++r) {
            int rowi = r + hi * 8;
            hid[rowi * HS + n]        = (_Float16)fmaxf(c0[r] + bias, 0.0f);
            hid[(16 + rowi) * HS + n] = (_Float16)fmaxf(c1[r] + bias, 0.0f);
        }
    }
    __syncthreads();

    if (dout_naive > 0) {
        // ---- layer2 VALU (tiny dout, e.g. node_dec dout=3) ----
        int total = MT * dout_naive;
        for (int q = t; q < total; q += 256) {
            int ml = q / dout_naive, j = q % dout_naive;
            int m = m0 + ml;
            if (m < M) {
                float acc = b2[j];
                for (int k = 0; k < HID; ++k)
                    acc += (float)hid[ml * HS + k] * w2_naive[k * dout_naive + j];
                out32[(long)m * dout_naive + j] = acc;
            }
        }
        return;
    }

    // ---- layer2 WMMA [32,128] x [128,128], shared B per wave ----
    v8f c0 = {}, c1 = {};
    const int n = wave * 16 + rloc;
    const _Float16* ar0  = hid + rloc * HS;
    const _Float16* ar1  = hid + (16 + rloc) * HS;
    const _Float16* brow = w2t + (long)n * HID;
#pragma unroll
    for (int k0 = 0; k0 < HID; k0 += 32) {
        v8h a0lo = *(const v8h*)(ar0 + k0 + b8);
        v8h a0hi = *(const v8h*)(ar0 + k0 + 16 + b8);
        v8h a1lo = *(const v8h*)(ar1 + k0 + b8);
        v8h a1hi = *(const v8h*)(ar1 + k0 + 16 + b8);
        v8h blo  = *(const v8h*)(brow + k0 + b8);
        v8h bhi  = *(const v8h*)(brow + k0 + 16 + b8);
        v16h a0 = __builtin_shufflevector(a0lo, a0hi, 0,1,2,3,4,5,6,7,8,9,10,11,12,13,14,15);
        v16h a1 = __builtin_shufflevector(a1lo, a1hi, 0,1,2,3,4,5,6,7,8,9,10,11,12,13,14,15);
        v16h b  = __builtin_shufflevector(blo,  bhi,  0,1,2,3,4,5,6,7,8,9,10,11,12,13,14,15);
        c0 = __builtin_amdgcn_wmma_f32_16x16x32_f16(false, a0, false, b, (short)0, c0, false, false);
        c1 = __builtin_amdgcn_wmma_f32_16x16x32_f16(false, a1, false, b, (short)0, c1, false, false);
    }
    const float bias = b2[n];
#pragma unroll
    for (int r = 0; r < 8; ++r) {
        int rowi = r + hi * 8;
#pragma unroll
        for (int half = 0; half < 2; ++half) {
            int rr = rowi + 16 * half;
            int m  = m0 + rr;
            if (m < M) {
                float v = (half ? c1[r] : c0[r]) + bias;
                if (res32) v += res32[(long)m * HID + n];
                if (out32) out32[(long)m * HID + n] = v;
                if (out16) out16[(long)m * HID + n] = (_Float16)v;
            }
        }
    }
}

// ---------------------------------------------------------------------------
// scatter-add: agg32[row[e], c] += e16[e, c]
// ---------------------------------------------------------------------------
__global__ void k_scatter_e(const _Float16* __restrict__ e16, const int* __restrict__ row,
                            long EH, float* __restrict__ agg32) {
    long tg = (long)blockIdx.x * 256 + threadIdx.x;
    if (tg < EH) {
        long eid = tg >> 7;
        int  c   = (int)(tg & 127);
        atomicAdd(&agg32[(long)row[eid] * HID + c], (float)e16[tg]);
    }
}

__global__ void k_agg_div(const float* __restrict__ agg32, const float* __restrict__ cnt,
                          long NH, _Float16* __restrict__ agg16) {
    long tg = (long)blockIdx.x * 256 + threadIdx.x;
    if (tg < NH)
        agg16[tg] = (_Float16)(agg32[tg] / fmaxf(cnt[tg >> 7], 1.0f));
}

// ---------------------------------------------------------------------------
// segment sums for graph-level means
// ---------------------------------------------------------------------------
__global__ void k_seg_node(const float* __restrict__ x32, const int* __restrict__ batch,
                           long NH, float* __restrict__ ng, float* __restrict__ ncnt) {
    long tg = (long)blockIdx.x * 256 + threadIdx.x;
    if (tg < NH) {
        long n = tg >> 7;
        int  c = (int)(tg & 127);
        int  b = batch[n];
        atomicAdd(&ng[b * HID + c], x32[tg]);
        if (c == 0) atomicAdd(&ncnt[b], 1.0f);
    }
}

__global__ void k_seg_edge(const _Float16* __restrict__ e16, const int* __restrict__ eb,
                           long EH, float* __restrict__ eg, float* __restrict__ ecnt) {
    long tg = (long)blockIdx.x * 256 + threadIdx.x;
    if (tg < EH) {
        long e = tg >> 7;
        int  c = (int)(tg & 127);
        int  b = eb[e];
        atomicAdd(&eg[b * HID + c], (float)e16[tg]);
        if (c == 0) atomicAdd(&ecnt[b], 1.0f);
    }
}

// ---------------------------------------------------------------------------
// head: means -> concat [ng, eg, u] (384) -> MLP(384->128->4). One block.
// ---------------------------------------------------------------------------
__global__ __launch_bounds__(128)
void k_head(const float* ng, const float* ncnt, const float* eg, const float* ecnt,
            const float* u32, const float* hw1, const float* hb1,
            const float* hw2, const float* hb2, int B, float* out) {
    __shared__ float hin[3 * HID];
    __shared__ float hh[HID];
    if (B > 8) B = 8;
    int j = threadIdx.x;
    for (int b = 0; b < B; ++b) {
        hin[j]           = ng[b * HID + j] / fmaxf(ncnt[b], 1.0f);
        hin[HID + j]     = eg[b * HID + j] / fmaxf(ecnt[b], 1.0f);
        hin[2 * HID + j] = u32[b * HID + j];
        __syncthreads();
        float a = hb1[j];
        for (int k = 0; k < 3 * HID; ++k) a += hin[k] * hw1[k * HID + j];
        hh[j] = fmaxf(a, 0.0f);
        __syncthreads();
        if (j < 4) {
            float o = hb2[j];
            for (int k = 0; k < HID; ++k) o += hh[k] * hw2[k * 4 + j];
            out[b * 4 + j] = o;
        }
        __syncthreads();
    }
}

// ---------------------------------------------------------------------------
// host orchestration
// ---------------------------------------------------------------------------
extern "C" void kernel_launch(void* const* d_in, const int* in_sizes, int n_in,
                              void* d_out, int out_size, void* d_ws, size_t ws_size,
                              hipStream_t stream) {
    const int N = in_sizes[0] / 6;
    const int E = in_sizes[1] / 4;
    const int B = in_sizes[2] / 16;
    const int NUM_LAYERS = 4;

    auto F = [&](int i) { return (const float*)d_in[i]; };
    // input index map (setup_inputs dict insertion order, params depth-first):
    // 0:x 1:edge_attr 2:conditions 3:scale
    // 4..7 node_enc(w1,b1,w2,b2)  8..11 edge_enc  12..15 cond_enc 16..19 scale_enc
    // 20..23 u_enc  24..27 node_dec  28..31 head
    // 32+8i..: layers[i].edge(w1,b1,w2,b2), layers[i].node(w1,b1,w2,b2)
    // 64: edge_index [2,E]  65: batch [N]
    const int* row   = (const int*)d_in[64];
    const int* col   = row + E;
    const int* batch = (const int*)d_in[65];

    // ---- carve workspace ----
    size_t off = 0;
    auto carve = [&](size_t bytes) -> void* {
        off = (off + 255) & ~(size_t)255;
        void* r = (char*)d_ws + off;
        off += bytes;
        return r;
    };
    float*     x32   = (float*)    carve((size_t)N * HID * 4);
    _Float16*  x16   = (_Float16*) carve((size_t)N * HID * 2);
    _Float16*  e16   = (_Float16*) carve((size_t)E * HID * 2);
    float*     agg32 = (float*)    carve((size_t)N * HID * 4);
    _Float16*  agg16 = (_Float16*) carve((size_t)N * HID * 2);
    float*     cnt   = (float*)    carve((size_t)N * 4);
    int*       eb    = (int*)      carve((size_t)E * 4);
    float*     u32   = (float*)    carve((size_t)8 * HID * 4);
    _Float16*  u16   = (_Float16*) carve((size_t)8 * HID * 2);
    float*     ng    = (float*)    carve((size_t)8 * HID * 4);
    float*     eg    = (float*)    carve((size_t)8 * HID * 4);
    float*     ncnt  = (float*)    carve(64);
    float*     ecnt  = (float*)    carve(64);
    _Float16*  wt_ne2  = (_Float16*) carve((size_t)HID * HID * 2);
    _Float16*  wt_ee2  = (_Float16*) carve((size_t)HID * HID * 2);
    _Float16*  wt_dec1 = (_Float16*) carve((size_t)HID * HID * 2);
    _Float16*  wt_e1[4]; _Float16* wt_e2[4]; _Float16* wt_n1[4]; _Float16* wt_n2[4];
    for (int i = 0; i < NUM_LAYERS; ++i) {
        wt_e1[i] = (_Float16*)carve((size_t)4 * HID * HID * 2);
        wt_e2[i] = (_Float16*)carve((size_t)HID * HID * 2);
        wt_n1[i] = (_Float16*)carve((size_t)3 * HID * HID * 2);
        wt_n2[i] = (_Float16*)carve((size_t)HID * HID * 2);
    }
    (void)ws_size;

    auto cdiv = [](long a, long b) { return (unsigned)((a + b - 1) / b); };

    // ---- weight convert + transpose (tiny; L2-resident afterwards) ----
    k_wt<<<cdiv((long)HID * HID, 256), 256, 0, stream>>>(F(6),  wt_ne2,  HID, HID);
    k_wt<<<cdiv((long)HID * HID, 256), 256, 0, stream>>>(F(10), wt_ee2,  HID, HID);
    k_wt<<<cdiv((long)HID * HID, 256), 256, 0, stream>>>(F(24), wt_dec1, HID, HID);
    for (int i = 0; i < NUM_LAYERS; ++i) {
        int b0 = 32 + 8 * i;
        k_wt<<<cdiv((long)4 * HID * HID, 256), 256, 0, stream>>>(F(b0 + 0), wt_e1[i], 4 * HID, HID);
        k_wt<<<cdiv((long)HID * HID, 256),     256, 0, stream>>>(F(b0 + 2), wt_e2[i], HID, HID);
        k_wt<<<cdiv((long)3 * HID * HID, 256), 256, 0, stream>>>(F(b0 + 4), wt_n1[i], 3 * HID, HID);
        k_wt<<<cdiv((long)HID * HID, 256),     256, 0, stream>>>(F(b0 + 6), wt_n2[i], HID, HID);
    }

    // ---- degree counts + edge_batch ----
    hipMemsetAsync(cnt, 0, (size_t)N * 4, stream);
    k_edge_batch_cnt<<<cdiv(E, 256), 256, 0, stream>>>(row, batch, E, eb, cnt);

    // ---- global (u) path ----
    k_u_path<<<1, 128, 0, stream>>>(F(2), F(3),
                                    F(12), F(13), F(14), F(15),
                                    F(16), F(17), F(18), F(19),
                                    F(20), F(21), F(22), F(23),
                                    B, u32, u16);

    // ---- node / edge encoders (thin layer1 + WMMA layer2) ----
    fused_mlp2<<<cdiv(N, MT), 256, 0, stream>>>(N,
        F(0), 6, F(4),
        nullptr, nullptr, nullptr, nullptr, nullptr, nullptr, nullptr, nullptr, 0,
        nullptr, F(5),
        wt_ne2, F(7), nullptr, 0,
        nullptr, x32, x16);
    fused_mlp2<<<cdiv(E, MT), 256, 0, stream>>>(E,
        F(1), 4, F(8),
        nullptr, nullptr, nullptr, nullptr, nullptr, nullptr, nullptr, nullptr, 0,
        nullptr, F(9),
        wt_ee2, F(11), nullptr, 0,
        nullptr, nullptr, e16);

    // ---- message-passing layers ----
    for (int i = 0; i < NUM_LAYERS; ++i) {
        int b0 = 32 + 8 * i;
        // edge MLP: concat(x[row], x[col], e, u[eb]) -> 512 -> 128 -> 128
        fused_mlp2<<<cdiv(E, MT), 256, 0, stream>>>(E,
            nullptr, 0, nullptr,
            x16, row, x16, col, e16, nullptr, u16, eb, 4,
            wt_e1[i], F(b0 + 1),
            wt_e2[i], F(b0 + 3), nullptr, 0,
            nullptr, nullptr, e16);
        // scatter-mean over row
        hipMemsetAsync(agg32, 0, (size_t)N * HID * 4, stream);
        k_scatter_e<<<cdiv((long)E * HID, 256), 256, 0, stream>>>(e16, row, (long)E * HID, agg32);
        k_agg_div<<<cdiv((long)N * HID, 256), 256, 0, stream>>>(agg32, cnt, (long)N * HID, agg16);
        // node MLP: concat(x, agg, u[batch]) -> 384 -> 128 -> 128, + residual
        fused_mlp2<<<cdiv(N, MT), 256, 0, stream>>>(N,
            nullptr, 0, nullptr,
            x16, nullptr, agg16, nullptr, u16, batch, nullptr, nullptr, 3,
            wt_n1[i], F(b0 + 5),
            wt_n2[i], F(b0 + 7), nullptr, 0,
            x32, x32, x16);
    }

    // ---- graph-level means ----
    hipMemsetAsync(ng, 0, (size_t)8 * HID * 4, stream);
    hipMemsetAsync(eg, 0, (size_t)8 * HID * 4, stream);
    hipMemsetAsync(ncnt, 0, 64, stream);
    hipMemsetAsync(ecnt, 0, 64, stream);
    k_seg_node<<<cdiv((long)N * HID, 256), 256, 0, stream>>>(x32, batch, (long)N * HID, ng, ncnt);
    k_seg_edge<<<cdiv((long)E * HID, 256), 256, 0, stream>>>(e16, eb, (long)E * HID, eg, ecnt);

    // ---- node decoder: 128 -> 128 (WMMA) -> 3 (VALU) straight into d_out ----
    fused_mlp2<<<cdiv(N, MT), 256, 0, stream>>>(N,
        nullptr, 0, nullptr,
        x16, nullptr, nullptr, nullptr, nullptr, nullptr, nullptr, nullptr, 1,
        wt_dec1, F(25),
        nullptr, F(27), F(26), 3,
        nullptr, (float*)d_out, nullptr);

    // ---- head -> log_ratios after x_out ----
    k_head<<<1, 128, 0, stream>>>(ng, ncnt, eg, ecnt, u32,
                                  F(28), F(29), F(30), F(31), B,
                                  (float*)d_out + (long)N * 3);
}
